// Decoder_17154099380367
// MI455X (gfx1250) — matrix operand order repaired and checked by reading, not compiled, not run on hardware
//
#include <hip/hip_runtime.h>
#include <hip/hip_bf16.h>
#include <stdint.h>

// ---- problem constants (match reference) ----
static constexpr int kS  = 2048;
static constexpr int kD  = 4096;
static constexpr int kH  = 32;
static constexpr int kHKV= 8;
static constexpr int kHD = 128;
static constexpr int kFF = 11008;
static constexpr float kGAMMA = 64.0f;
static constexpr float kEPS   = 1e-5f;

typedef __attribute__((ext_vector_type(8)))  __bf16 v8bf;
typedef __attribute__((ext_vector_type(16))) __bf16 v16bf;
typedef __attribute__((ext_vector_type(8)))  float  v8fx;

enum { EPI_NONE = 0, EPI_RES = 1, EPI_SAT = 2 };

// =====================================================================
// One-time weight prep: f32 [K,N] -> bf16 [N,K] (transpose), batched.
// =====================================================================
__global__ __launch_bounds__(256)
void convert_transpose_kernel(const float* __restrict__ in, __bf16* __restrict__ out,
                              int K, int N)
{
    const long long base = (long long)blockIdx.z * K * N;
    const long long id = (long long)blockIdx.x * 256 + threadIdx.x;
    const long long total = (long long)K * N;
    if (id >= total) return;
    const int n = (int)(id / K);
    const int k = (int)(id % K);
    out[base + id] = (__bf16)in[base + (long long)k * N + n];
}

// =====================================================================
// Batched/strided WMMA GEMM, async double-buffered edition.
//   C[z](m,n) = alpha * sum_k A[z][m*lda+k] * B[z][n*ldb+k]  (+Res / sat)
//   Tile: BM=128, BN=64, BK=32; 256 thr = 8 waves (4x2), wave tile 32x32.
//   LDS staged with GLOBAL_LOAD_ASYNC_TO_LDS_B128 (ASYNCcnt), ping-pong
//   buffers: issue k+1, s_wait_asynccnt 3 -> tile k landed, compute.
//   4x v_wmma_f32_16x16x32_bf16 per wave per K-step with reuse hints.
// =====================================================================
__global__ __launch_bounds__(256)
void gemm_bf16_wmma_async(const __bf16* __restrict__ A, int lda, long long strideA, int azdiv,
                          const __bf16* __restrict__ B, int ldb, long long strideB, int bzdiv,
                          void* __restrict__ Cv, int ldc, long long strideC, int cbf,
                          const float* __restrict__ Res,
                          int M, int N, int K, float alpha, int epi)
{
    __shared__ __bf16 Alds[2][128 * 32];   // [buf][row][k]
    __shared__ __bf16 Blds[2][64 * 32];    // [buf][n][k]

    const int z = blockIdx.z;
    A += (long long)(z / azdiv) * strideA;
    B += (long long)(z / bzdiv) * strideB;
    const float* R = Res ? (Res + (long long)z * strideC) : nullptr;

    const int m0 = blockIdx.y * 128;
    const int n0 = blockIdx.x * 64;
    const int tid  = threadIdx.x;
    const int lane = tid & 31;
    const int wave = tid >> 5;
    const int wm = wave & 3;          // 4 wave-rows of 32
    const int wn = wave >> 2;         // 2 wave-cols of 32
    const int l15  = lane & 15;
    const int half = lane >> 4;

    // ---- async copy slots ----
    // A tile: 128 rows x 64B; 2 threads/row, 2x16B ops per thread
    const int arowi = tid >> 1;
    const int aseg  = (tid & 1) * 8;             // elements
    const __bf16* Ag = A + (long long)(m0 + arowi) * lda + aseg;
    const unsigned aoff0[2] = { (unsigned)(uintptr_t)&Alds[0][arowi * 32 + aseg],
                                (unsigned)(uintptr_t)&Alds[1][arowi * 32 + aseg] };
    // B tile: 64 rows x 64B; 4 threads/row, 1x16B op per thread
    const int browi = tid >> 2;
    const int bseg  = (tid & 3) * 8;
    const __bf16* Bg = B + (long long)(n0 + browi) * ldb + bseg;
    const unsigned boff[2] = { (unsigned)(uintptr_t)&Blds[0][browi * 32 + bseg],
                               (unsigned)(uintptr_t)&Blds[1][browi * 32 + bseg] };

    auto issue_tile = [&](int kt, int buf) {
        const unsigned long long ga0 = (unsigned long long)(uintptr_t)(Ag + kt * 32);
        const unsigned long long ga1 = ga0 + 32;                 // +16 elements
        const unsigned long long gb  = (unsigned long long)(uintptr_t)(Bg + kt * 32);
        asm volatile("global_load_async_to_lds_b128 %0, %1, off"
                     :: "v"(aoff0[buf]), "v"(ga0) : "memory");
        asm volatile("global_load_async_to_lds_b128 %0, %1, off"
                     :: "v"(aoff0[buf] + 32u), "v"(ga1) : "memory");
        asm volatile("global_load_async_to_lds_b128 %0, %1, off"
                     :: "v"(boff[buf]), "v"(gb) : "memory");
    };

    v8fx c00 = {}, c01 = {}, c10 = {}, c11 = {};
    const int nk = K >> 5;

    issue_tile(0, 0);
    for (int i = 0; i < nk; ++i) {
        const int cur = i & 1;
        if (i + 1 < nk) {
            issue_tile(i + 1, cur ^ 1);
            __builtin_prefetch(Ag + (i + 2) * 32, 0, 1);   // global_prefetch_b8 -> GL2
            __builtin_prefetch(Bg + (i + 2) * 32, 0, 1);
            asm volatile("s_wait_asynccnt 0x3" ::: "memory");   // tile i landed (in-order)
        } else {
            asm volatile("s_wait_asynccnt 0x0" ::: "memory");
        }
        __syncthreads();

        const __bf16* Ab = &Alds[cur][0];
        const __bf16* Bb = &Blds[cur][0];

        // ---- two A fragments (rows wm*32 + {0,16}), 16-bit A layout ----
        v16bf a0, a1;
        {
            const int r0 = (wm * 32 + l15) * 32;
            const v8bf lo0 = *(const v8bf*)&Ab[r0 + half * 8];
            const v8bf hi0 = *(const v8bf*)&Ab[r0 + half * 8 + 16];
            const int r1 = (wm * 32 + 16 + l15) * 32;
            const v8bf lo1 = *(const v8bf*)&Ab[r1 + half * 8];
            const v8bf hi1 = *(const v8bf*)&Ab[r1 + half * 8 + 16];
            #pragma unroll
            for (int t = 0; t < 8; ++t) {
                a0[t] = lo0[t]; a0[t + 8] = hi0[t];
                a1[t] = lo1[t]; a1[t + 8] = hi1[t];
            }
        }
        // ---- two B fragments (cols wn*32 + {0,16}) ----
        v16bf b0, b1;
        {
            const int q0 = (wn * 32 + l15) * 32;
            const v8bf lo0 = *(const v8bf*)&Bb[q0 + half * 8];
            const v8bf hi0 = *(const v8bf*)&Bb[q0 + half * 8 + 16];
            const int q1 = (wn * 32 + 16 + l15) * 32;
            const v8bf lo1 = *(const v8bf*)&Bb[q1 + half * 8];
            const v8bf hi1 = *(const v8bf*)&Bb[q1 + half * 8 + 16];
            #pragma unroll
            for (int t = 0; t < 8; ++t) {
                b0[t] = lo0[t]; b0[t + 8] = hi0[t];
                b1[t] = lo1[t]; b1[t + 8] = hi1[t];
            }
        }
        // snake order with operand-reuse hints (VOP3P RA/RB)
        c00 = __builtin_amdgcn_wmma_f32_16x16x32_bf16(false, a0, false, b0, (short)0, c00, true,  false);
        c01 = __builtin_amdgcn_wmma_f32_16x16x32_bf16(false, a0, false, b1, (short)0, c01, false, true );
        c11 = __builtin_amdgcn_wmma_f32_16x16x32_bf16(false, a1, false, b1, (short)0, c11, true,  false);
        c10 = __builtin_amdgcn_wmma_f32_16x16x32_bf16(false, a1, false, b0, (short)0, c10, false, false);
        __syncthreads();
    }

    // ---- epilogue: 16x16 f32 C/D layout (VGPR j -> M = j + half*8) ----
    float* Cf = (float*)Cv + (long long)z * strideC;
    __bf16* Cb = (__bf16*)Cv + (long long)z * strideC;
    #pragma unroll
    for (int mf = 0; mf < 2; ++mf) {
        #pragma unroll
        for (int nf = 0; nf < 2; ++nf) {
            const v8fx cc = mf ? (nf ? c11 : c10) : (nf ? c01 : c00);
            const int col = n0 + wn * 32 + nf * 16 + l15;
            #pragma unroll
            for (int j = 0; j < 8; ++j) {
                const int row = m0 + wm * 32 + mf * 16 + half * 8 + j;
                float v = alpha * cc[j];
                if (epi == EPI_RES)      v += R[(long long)row * ldc + col];
                else if (epi == EPI_SAT) v = v / (1.0f + fabsf(v));
                const long long at = (long long)row * ldc + col;
                if (cbf) Cb[at] = (__bf16)v;
                else     Cf[at] = v;
            }
        }
    }
}

// =====================================================================
// RMSNorm: f32 in, bf16 out
// =====================================================================
__global__ __launch_bounds__(256)
void rmsnorm_kernel(const float* __restrict__ x, const float* __restrict__ w,
                    __bf16* __restrict__ o)
{
    __shared__ float red[8];
    const int s = blockIdx.x;
    const float* xr = x + (long long)s * kD;
    float acc = 0.f;
    for (int i = threadIdx.x; i < kD; i += 256) { float v = xr[i]; acc += v * v; }
    #pragma unroll
    for (int off = 16; off; off >>= 1) acc += __shfl_down(acc, off, 32);
    if ((threadIdx.x & 31) == 0) red[threadIdx.x >> 5] = acc;
    __syncthreads();
    if (threadIdx.x == 0) {
        float t = 0.f;
        #pragma unroll
        for (int i = 0; i < 8; ++i) t += red[i];
        red[0] = rsqrtf(t / (float)kD + kEPS);
    }
    __syncthreads();
    const float rs = red[0];
    __bf16* orow = o + (long long)s * kD;
    for (int i = threadIdx.x; i < kD; i += 256) orow[i] = (__bf16)(xr[i] * rs * w[i]);
}

// =====================================================================
// RoPE: f32 [S,nh,128] in -> rotated bf16 out (same layout)
// =====================================================================
__global__ __launch_bounds__(256)
void rope_kernel(const float* __restrict__ x, __bf16* __restrict__ o, int nh)
{
    const long long id = (long long)blockIdx.x * 256 + threadIdx.x;
    const long long total = (long long)kS * nh * 64;
    if (id >= total) return;
    const int j = (int)(id & 63);
    long long r = id >> 6;
    const int h = (int)(r % nh);
    const long long s = r / nh;
    const float inv = __expf(-(2.0f * (float)j / 128.0f) * 9.210340371976184f);
    const float ang = (float)s * inv;
    float sn, cs;
    __sincosf(ang, &sn, &cs);
    const long long base = ((long long)s * nh + h) * 128 + j;
    const float x0 = x[base], x1 = x[base + 64];
    o[base]      = (__bf16)(x0 * cs - x1 * sn);
    o[base + 64] = (__bf16)(x1 * cs + x0 * sn);
}

// =====================================================================
// Causal online-softmax * V.  One wave per (head, query row).
// float4 score reads (4 keys/iter); bf16 output for the Wo GEMM.
// =====================================================================
__global__ __launch_bounds__(256)
void softmax_av_kernel(const float* __restrict__ scores, const float* __restrict__ V,
                       __bf16* __restrict__ out)
{
    const int wid  = blockIdx.x * 8 + (threadIdx.x >> 5);
    const int lane = threadIdx.x & 31;
    const int h = wid / kS;
    const int s = wid % kS;
    const float* srow = scores + ((long long)h * kS + s) * kS;
    const float* vbase = V + (long long)(h / (kH / kHKV)) * kHD;
    float m = -3.4e38f, l = 0.f;
    float acc[4] = {0.f, 0.f, 0.f, 0.f};

    const int len = s + 1;
    const int len4 = len & ~3;
    for (int t = 0; t < len4; t += 4) {
        const float4 xq = *(const float4*)&srow[t];
        const float xv[4] = { xq.x, xq.y, xq.z, xq.w };
        #pragma unroll
        for (int u = 0; u < 4; ++u) {
            const float mn = fmaxf(m, xv[u]);
            const float corr = __expf(m - mn);
            const float p = __expf(xv[u] - mn);
            l = l * corr + p;
            const float* vr = vbase + (long long)(t + u) * (kHKV * kHD);
            #pragma unroll
            for (int i = 0; i < 4; ++i) acc[i] = acc[i] * corr + p * vr[lane + 32 * i];
            m = mn;
        }
    }
    for (int t = len4; t < len; ++t) {
        const float xv = srow[t];
        const float mn = fmaxf(m, xv);
        const float corr = __expf(m - mn);
        const float p = __expf(xv - mn);
        l = l * corr + p;
        const float* vr = vbase + (long long)t * (kHKV * kHD);
        #pragma unroll
        for (int i = 0; i < 4; ++i) acc[i] = acc[i] * corr + p * vr[lane + 32 * i];
        m = mn;
    }
    const float inv = 1.0f / l;
    __bf16* orow = out + (long long)s * (kH * kHD) + (long long)h * kHD;
    #pragma unroll
    for (int i = 0; i < 4; ++i) orow[lane + 32 * i] = (__bf16)(acc[i] * inv);
}

// =====================================================================
// act = silu(gate) * up   (bf16 in/out, in place into gate)
// =====================================================================
__global__ __launch_bounds__(256)
void silu_mul_kernel(__bf16* __restrict__ g, const __bf16* __restrict__ u, long long n)
{
    const long long i = (long long)blockIdx.x * 256 + threadIdx.x;
    if (i < n) {
        const float a = (float)g[i];
        g[i] = (__bf16)((a / (1.0f + __expf(-a))) * (float)u[i]);
    }
}

// =====================================================================
// Host-side orchestration
// =====================================================================
extern "C" void kernel_launch(void* const* d_in, const int* in_sizes, int n_in,
                              void* d_out, int out_size, void* d_ws, size_t ws_size,
                              hipStream_t stream)
{
    (void)in_sizes; (void)n_in; (void)out_size; (void)ws_size;

    const float* hs   = (const float*)d_in[0];
    const float* ln1w = (const float*)d_in[1];
    const float* ln2w = (const float*)d_in[2];
    const float* Wq   = (const float*)d_in[3];
    const float* Wk   = (const float*)d_in[4];
    const float* Wv   = (const float*)d_in[5];
    const float* Wo   = (const float*)d_in[6];
    const float* rot  = (const float*)d_in[7];
    const float* Wg   = (const float*)d_in[8];
    const float* Wu   = (const float*)d_in[9];
    const float* Wd   = (const float*)d_in[10];

    float* out = (float*)d_out;
    const long long SD  = (long long)kS * kD;
    const long long SF  = (long long)kS * kFF;
    const long long HSS = (long long)kH * kS * kS;

    float* out_hidden = out;            // [S, D]
    float* out_draft  = out + SD;       // [H, S, S]
    float* out_true   = out + SD + HSS; // [H, S, S]

    // ---- bump allocator over d_ws (256B aligned chunks) ----
    char* wp = (char*)d_ws;
    auto alloc = [&](size_t bytes) -> void* {
        void* r = (void*)wp;
        wp += (bytes + 255) & ~(size_t)255;
        return r;
    };
    __bf16* WqT  = (__bf16*)alloc((size_t)kD * kD * 2);
    __bf16* WkT  = (__bf16*)alloc((size_t)kHKV * kHD * kD * 2);
    __bf16* WvT  = (__bf16*)alloc((size_t)kHKV * kHD * kD * 2);
    __bf16* WoT  = (__bf16*)alloc((size_t)kD * kD * 2);
    __bf16* rotT = (__bf16*)alloc((size_t)kH * kHD * kHD * 2);
    __bf16* WgT  = (__bf16*)alloc((size_t)kFF * kD * 2);
    __bf16* WuT  = (__bf16*)alloc((size_t)kFF * kD * 2);
    __bf16* WdT  = (__bf16*)alloc((size_t)kD * kFF * 2);
    __bf16* xn   = (__bf16*)alloc((size_t)SD * 2);
    float*  qb   = (float*) alloc((size_t)SD * 4);
    float*  kb   = (float*) alloc((size_t)kS * kHKV * kHD * 4);
    float*  vb   = (float*) alloc((size_t)kS * kHKV * kHD * 4);
    __bf16* qbh  = (__bf16*)alloc((size_t)SD * 2);
    __bf16* kbh  = (__bf16*)alloc((size_t)kS * kHKV * kHD * 2);
    __bf16* qh   = (__bf16*)alloc((size_t)kH * kS * kHD * 2);
    __bf16* kh   = (__bf16*)alloc((size_t)kH * kS * kHD * 2);
    __bf16* attn = (__bf16*)alloc((size_t)SD * 2);
    float*  hid  = (float*) alloc((size_t)SD * 4);
    __bf16* gate = (__bf16*)alloc((size_t)SF * 2);
    __bf16* up   = (__bf16*)alloc((size_t)SF * 2);

    const dim3 blk(256);
    const float inv_sqrt_hd = 0.08838834764831845f;
    auto cgrid = [](long long n, int batch) { return dim3((unsigned)((n + 255) / 256), 1, batch); };

    // 0) one-time weight convert+transpose to bf16 N-major
    convert_transpose_kernel<<<cgrid((long long)kD * kD, 1), blk, 0, stream>>>(Wq, WqT, kD, kD);
    convert_transpose_kernel<<<cgrid((long long)kD * kHKV * kHD, 1), blk, 0, stream>>>(Wk, WkT, kD, kHKV * kHD);
    convert_transpose_kernel<<<cgrid((long long)kD * kHKV * kHD, 1), blk, 0, stream>>>(Wv, WvT, kD, kHKV * kHD);
    convert_transpose_kernel<<<cgrid((long long)kD * kD, 1), blk, 0, stream>>>(Wo, WoT, kD, kD);
    convert_transpose_kernel<<<cgrid((long long)kHD * kHD, kH), blk, 0, stream>>>(rot, rotT, kHD, kHD);
    convert_transpose_kernel<<<cgrid((long long)kD * kFF, 1), blk, 0, stream>>>(Wg, WgT, kD, kFF);
    convert_transpose_kernel<<<cgrid((long long)kD * kFF, 1), blk, 0, stream>>>(Wu, WuT, kD, kFF);
    convert_transpose_kernel<<<cgrid((long long)kFF * kD, 1), blk, 0, stream>>>(Wd, WdT, kFF, kD);

    // 1) RMSNorm #1 -> bf16
    rmsnorm_kernel<<<dim3(kS), blk, 0, stream>>>(hs, ln1w, xn);

    // 2) Q/K/V projections
    gemm_bf16_wmma_async<<<dim3(kD / 64, kS / 128, 1), blk, 0, stream>>>(
        xn, kD, 0, 1, WqT, kD, 0, 1, qb, kD, 0, 0, nullptr,
        kS, kD, kD, 1.0f, EPI_NONE);
    gemm_bf16_wmma_async<<<dim3((kHKV * kHD) / 64, kS / 128, 1), blk, 0, stream>>>(
        xn, kD, 0, 1, WkT, kD, 0, 1, kb, kHKV * kHD, 0, 0, nullptr,
        kS, kHKV * kHD, kD, 1.0f, EPI_NONE);
    gemm_bf16_wmma_async<<<dim3((kHKV * kHD) / 64, kS / 128, 1), blk, 0, stream>>>(
        xn, kD, 0, 1, WvT, kD, 0, 1, vb, kHKV * kHD, 0, 0, nullptr,
        kS, kHKV * kHD, kD, 1.0f, EPI_NONE);

    // 3) RoPE -> bf16
    rope_kernel<<<cgrid((long long)kS * kH * 64, 1), blk, 0, stream>>>(qb, qbh, kH);
    rope_kernel<<<cgrid((long long)kS * kHKV * 64, 1), blk, 0, stream>>>(kb, kbh, kHKV);

    // 4) per-head hash projections: y = sat(GAMMA * x @ rot[h]) -> bf16
    gemm_bf16_wmma_async<<<dim3(kHD / 64, kS / 128, kH), blk, 0, stream>>>(
        qbh, kH * kHD, (long long)kHD, 1,
        rotT, kHD, (long long)kHD * kHD, 1,
        qh, kHD, (long long)kS * kHD, 1, nullptr,
        kS, kHD, kHD, kGAMMA, EPI_SAT);
    gemm_bf16_wmma_async<<<dim3(kHD / 64, kS / 128, kH), blk, 0, stream>>>(
        kbh, kHKV * kHD, (long long)kHD, kH / kHKV,
        rotT, kHD, (long long)kHD * kHD, 1,
        kh, kHD, (long long)kS * kHD, 1, nullptr,
        kS, kHD, kHD, kGAMMA, EPI_SAT);

    // 5) draft_attn[h] = qh[h] @ kh[h]^T -> d_out f32
    gemm_bf16_wmma_async<<<dim3(kS / 64, kS / 128, kH), blk, 0, stream>>>(
        qh, kHD, (long long)kS * kHD, 1,
        kh, kHD, (long long)kS * kHD, 1,
        out_draft, kS, (long long)kS * kS, 0, nullptr,
        kS, kS, kHD, 1.0f, EPI_NONE);

    // 6) true_attn[h] = (q_r[h] @ k_r[h/4]^T) / sqrt(HD) -> d_out f32 (unmasked)
    gemm_bf16_wmma_async<<<dim3(kS / 64, kS / 128, kH), blk, 0, stream>>>(
        qbh, kH * kHD, (long long)kHD, 1,
        kbh, kHKV * kHD, (long long)kHD, kH / kHKV,
        out_true, kS, (long long)kS * kS, 0, nullptr,
        kS, kS, kHD, inv_sqrt_hd, EPI_NONE);

    // 7) causal softmax(true_attn) @ V -> attn bf16
    softmax_av_kernel<<<dim3((kH * kS) / 8), blk, 0, stream>>>(out_true, vb, attn);

    // 8) output projection + residual
    gemm_bf16_wmma_async<<<dim3(kD / 64, kS / 128, 1), blk, 0, stream>>>(
        attn, kH * kHD, 0, 1, WoT, kD, 0, 1, hid, kD, 0, 0, hs,
        kS, kD, kH * kHD, 1.0f, EPI_RES);

    // 9) RMSNorm #2 -> bf16 (reuse xn)
    rmsnorm_kernel<<<dim3(kS), blk, 0, stream>>>(hid, ln2w, xn);

    // 10) MLP gate/up -> bf16
    gemm_bf16_wmma_async<<<dim3(kFF / 64, kS / 128, 1), blk, 0, stream>>>(
        xn, kD, 0, 1, WgT, kD, 0, 1, gate, kFF, 0, 1, nullptr,
        kS, kFF, kD, 1.0f, EPI_NONE);
    gemm_bf16_wmma_async<<<dim3(kFF / 64, kS / 128, 1), blk, 0, stream>>>(
        xn, kD, 0, 1, WuT, kD, 0, 1, up, kFF, 0, 1, nullptr,
        kS, kFF, kD, 1.0f, EPI_NONE);

    // 11) gate = silu(gate) * up
    silu_mul_kernel<<<cgrid(SF, 1), blk, 0, stream>>>(gate, up, SF);

    // 12) down projection + residual -> final hidden output
    gemm_bf16_wmma_async<<<dim3(kD / 64, kS / 128, 1), blk, 0, stream>>>(
        gate, kFF, 0, 1, WdT, kFF, 0, 1, out_hidden, kD, 0, 0, hid,
        kS, kD, kFF, 1.0f, EPI_RES);
}
